// MultiHeadCausalSelfAttention_4423816315325
// MI455X (gfx1250) — matrix operand order repaired and checked by reading, not compiled, not run on hardware
//
#include <hip/hip_runtime.h>

// MI455X / gfx1250: wave32, WMMA bf16 16x16x32 (fp32 accum), async global->LDS
// double-buffered pipeline. B=4, T=2048, D=1024, H=16, DH=64.

typedef __attribute__((ext_vector_type(16))) __bf16 v16bf;
typedef __attribute__((ext_vector_type(8)))  float  v8f;
typedef __attribute__((ext_vector_type(4)))  int    v4i;

#if defined(__has_builtin)
#if __has_builtin(__builtin_amdgcn_global_load_async_to_lds_b128)
#define HAVE_ASYNC 1
#endif
#endif
#ifndef HAVE_ASYNC
#define HAVE_ASYNC 0
#endif

__device__ __forceinline__ unsigned short f2bf(float f) {
    union { float f; unsigned u; } v; v.f = f;
    unsigned r = v.u + 0x7FFFu + ((v.u >> 16) & 1u);   // round-to-nearest-even
    return (unsigned short)(r >> 16);
}

__device__ __forceinline__ v8f v8f_zero() {
    v8f z = {0.f,0.f,0.f,0.f,0.f,0.f,0.f,0.f};
    return z;
}

// 16-byte global->LDS copy: async path (ASYNCcnt-tracked) when available, else sync.
__device__ __forceinline__ void copy16(const unsigned short* g, unsigned short* l) {
#if HAVE_ASYNC
    __builtin_amdgcn_global_load_async_to_lds_b128(
        (__attribute__((address_space(1))) v4i*)g,
        (__attribute__((address_space(3))) v4i*)l, 0, 0);
#else
    *(uint4*)l = *(const uint4*)g;
#endif
}

__device__ __forceinline__ void wait_async() {
#if HAVE_ASYNC
#if __has_builtin(__builtin_amdgcn_s_wait_asynccnt)
    __builtin_amdgcn_s_wait_asynccnt(0);
#else
    asm volatile("s_wait_asynccnt 0x0" ::: "memory");
#endif
#endif
}

union FragU { v16bf v; uint4 q[2]; };

// A-fragment (16x32 bf16): lane l holds row m=l%16, K chunks {g*8..+7, g*8+16..+23}, g=l/16.
__device__ __forceinline__ v16bf load_frag_a(const unsigned short* lds, int row, int stride,
                                             int g, int koff) {
    FragU f;
    const int base = row * stride + koff + g * 8;
    f.q[0] = *(const uint4*)(lds + base);
    f.q[1] = *(const uint4*)(lds + base + 16);
    return f.v;
}

// B-fragment (32x16 bf16), LDS [n][k]: lane l holds col n=l%16, K chunks {g*8..+7, g*8+16..+23}, g=l/16.
__device__ __forceinline__ v16bf load_frag_b(const unsigned short* lds, int row, int stride,
                                             int g, int koff) {
    FragU f;
    const int base = row * stride + koff + g * 8;
    f.q[0] = *(const uint4*)(lds + base);
    f.q[1] = *(const uint4*)(lds + base + 16);
    return f.v;
}

// ---------------------------------------------------------------- fp32 -> bf16 (row-major copy)
__global__ __launch_bounds__(256) void cvt_f32_bf16(const float* __restrict__ in,
                                                    unsigned short* __restrict__ out) {
    size_t i = ((size_t)blockIdx.x * 256 + threadIdx.x) * 8;
    float4 a = *(const float4*)(in + i);
    float4 b = *(const float4*)(in + i + 4);
    alignas(16) unsigned short o[8] = { f2bf(a.x), f2bf(a.y), f2bf(a.z), f2bf(a.w),
                                        f2bf(b.x), f2bf(b.y), f2bf(b.z), f2bf(b.w) };
    *(uint4*)(out + i) = *(const uint4*)o;
}

// ---------------------------------------------------------------- fp32 [R][C] -> bf16 [C][R]
// One-time weight transpose so the GEMM B-operand is [N][K]: hot loops then have zero transposes.
__global__ __launch_bounds__(256) void cvt_transpose_bf16(const float* __restrict__ in,
                                                          unsigned short* __restrict__ out,
                                                          int R, int C) {
    __shared__ float t[32][33];
    const int tid = threadIdx.x;
    const int r0 = blockIdx.y * 32, c0 = blockIdx.x * 32;
    for (int i = 0; i < 4; ++i) {
        int r = (tid >> 5) + i * 8, c = tid & 31;
        t[r][c] = in[(size_t)(r0 + r) * C + c0 + c];
    }
    __syncthreads();
    for (int i = 0; i < 4; ++i) {
        int cr = (tid >> 5) + i * 8, cc = tid & 31;
        out[(size_t)(c0 + cr) * R + r0 + cc] = f2bf(t[cc][cr]);
    }
}

// ---------------------------------------------------------------- bf16 GEMM (B pre-transposed)
// C[M,N] = A[M,K] * Bt[N,K]^T.  128x128x64 tiles, double-buffered async LDS pipeline,
// 8 waves (2x4), 64x32 per wave => 16 WMMAs per wave per tile.
// epi==0: fp32 store to Cf.  epi==1: QKV split -> Q,K bf16 [B,H,T,64] (Q*0.125), V bf16 [B,H,64,T].
__global__ __launch_bounds__(256) void gemm_bf16(const unsigned short* __restrict__ A,
                                                 const unsigned short* __restrict__ Bt,
                                                 float* __restrict__ Cf,
                                                 unsigned short* __restrict__ Qo,
                                                 unsigned short* __restrict__ Ko,
                                                 unsigned short* __restrict__ Vo,
                                                 int N, int K, int epi) {
    __shared__ unsigned short ldsA[2 * 128 * 72];   // [buf][m][k], stride 72 (16B aligned rows)
    __shared__ unsigned short ldsB[2 * 128 * 72];   // [buf][n][k]
    const int tid  = threadIdx.x;
    const int lane = tid & 31, w = tid >> 5;
    const int wm = w >> 2, wn = w & 3;
    const int g = lane >> 4, ln = lane & 15;
    const int m0 = blockIdx.y * 128, n0 = blockIdx.x * 128;

    v8f acc[4][2];
    for (int mt = 0; mt < 4; ++mt)
        for (int nt = 0; nt < 2; ++nt) acc[mt][nt] = v8f_zero();

    auto issue_tile = [&](int kt, int buf) {
        unsigned short* la = ldsA + buf * (128 * 72);
        unsigned short* lb = ldsB + buf * (128 * 72);
        for (int it = 0; it < 4; ++it) {
            int c = tid + it * 256;
            int m = c >> 3, ch = c & 7;               // 128 rows x 8 x 16B chunks
            copy16(A  + (size_t)(m0 + m) * K + kt + ch * 8, la + m * 72 + ch * 8);
            copy16(Bt + (size_t)(n0 + m) * K + kt + ch * 8, lb + m * 72 + ch * 8);
        }
    };

    const int nk = K / 64;
    issue_tile(0, 0);
    for (int i = 0; i < nk; ++i) {
        wait_async();
        __syncthreads();
        if (i + 1 < nk) issue_tile((i + 1) * 64, (i + 1) & 1);  // overlap copy(i+1) with compute(i)
        const unsigned short* la = ldsA + (i & 1) * (128 * 72);
        const unsigned short* lb = ldsB + (i & 1) * (128 * 72);
        for (int ks = 0; ks < 2; ++ks) {
            v16bf af[4], bfg[2];
            for (int mt = 0; mt < 4; ++mt)
                af[mt] = load_frag_a(la, wm * 64 + mt * 16 + ln, 72, g, ks * 32);
            for (int nt = 0; nt < 2; ++nt)
                bfg[nt] = load_frag_b(lb, wn * 32 + nt * 16 + ln, 72, g, ks * 32);
            for (int mt = 0; mt < 4; ++mt)
                for (int nt = 0; nt < 2; ++nt)
                    acc[mt][nt] = __builtin_amdgcn_wmma_f32_16x16x32_bf16(
                        false, af[mt], false, bfg[nt], (short)0, acc[mt][nt], false, false);
        }
    }

    for (int mt = 0; mt < 4; ++mt)
        for (int nt = 0; nt < 2; ++nt)
            for (int r = 0; r < 8; ++r) {
                int row = m0 + wm * 64 + mt * 16 + r + (g << 3);
                int col = n0 + wn * 32 + nt * 16 + ln;
                float v = acc[mt][nt][r];
                if (epi == 0) {
                    Cf[(size_t)row * N + col] = v;
                } else {
                    int which = col >> 10, d = col & 1023;
                    int h = d >> 6, dh = d & 63;
                    int b = row >> 11, t = row & 2047;
                    if (which == 2) {   // V stored transposed [B,H,64,T] for transpose-free attn
                        Vo[((size_t)(b * 16 + h) * 64 + dh) * 2048 + t] = f2bf(v);
                    } else {
                        size_t idx = ((size_t)(b * 16 + h) * 2048 + t) * 64 + dh;
                        unsigned short bv = f2bf(which == 0 ? v * 0.125f : v); // 1/sqrt(64)
                        (which == 0 ? Qo : Ko)[idx] = bv;
                    }
                }
            }
}

// ---------------------------------------------------------------- causal flash attention
// Q,K bf16 [B*H, T, 64] (Q pre-scaled), V bf16 [B*H, 64, T]. Out bf16 [B*T, D] heads merged.
// One block per (b*h, 128 queries); wave owns 16 queries; 32 keys/step, double-buffered async K/V.
__global__ __launch_bounds__(256) void flash_attn(const unsigned short* __restrict__ Qb,
                                                  const unsigned short* __restrict__ Kb,
                                                  const unsigned short* __restrict__ VT,
                                                  unsigned short* __restrict__ Ob) {
    __shared__ unsigned short ldsK[2 * 32 * 72];      // [buf][key][dh]
    __shared__ unsigned short ldsV[2 * 64 * 40];      // [buf][dh][key]
    __shared__ unsigned short ldsP[8 * 16 * 40];      // per-wave P scratch [m][key]
    const int tid = threadIdx.x, lane = tid & 31, w = tid >> 5;
    const int g = lane >> 4, ln = lane & 15;
    const int qblock = blockIdx.x, bh = blockIdx.y;
    const int b = bh >> 4, h = bh & 15;
    const int qbase = qblock * 128 + w * 16;
    const size_t head_off = (size_t)bh * 2048 * 64;
    const size_t vt_off   = (size_t)bh * 64 * 2048;

    v16bf qf[2];
    {
        const unsigned short* qp = Qb + head_off + (size_t)(qbase + ln) * 64;
        FragU f;
        for (int fi = 0; fi < 2; ++fi) {
            f.q[0] = *(const uint4*)(qp + fi * 32 + g * 8);
            f.q[1] = *(const uint4*)(qp + fi * 32 + g * 8 + 16);
            qf[fi] = f.v;
        }
    }

    v8f oacc[4];
    for (int t = 0; t < 4; ++t) oacc[t] = v8f_zero();
    float Mr[8], Lr[8];
    for (int r = 0; r < 8; ++r) { Mr[r] = -1e30f; Lr[r] = 0.f; }
    unsigned short* myP = ldsP + w * 640;

    auto issue_kv = [&](int kt, int buf) {
        unsigned short* lk = ldsK + buf * (32 * 72);
        unsigned short* lv = ldsV + buf * (64 * 40);
        { int row = tid >> 3, ch = tid & 7;   // 32 keys x 8 chunks
          copy16(Kb + head_off + (size_t)(kt + row) * 64 + ch * 8, lk + row * 72 + ch * 8); }
        { int dh = tid >> 2, ch = tid & 3;    // 64 dh rows x 4 chunks of 8 keys
          copy16(VT + vt_off + (size_t)dh * 2048 + kt + ch * 8, lv + dh * 40 + ch * 8); }
    };

    const int jmax = qblock * 4 + 3;
    issue_kv(0, 0);
    for (int j = 0; j <= jmax; ++j) {
        const int kt = j * 32;
        wait_async();
        __syncthreads();
        if (j < jmax) issue_kv(kt + 32, (j + 1) & 1);
        const unsigned short* lk = ldsK + (j & 1) * (32 * 72);
        const unsigned short* lv = ldsV + (j & 1) * (64 * 40);

        // S = Q * K^T : two 16-key tiles, dh=64 -> 2 WMMAs each
        v8f s0 = v8f_zero(), s1 = v8f_zero();
        for (int fi = 0; fi < 2; ++fi) {
            v16bf bk0 = load_frag_b(lk, ln,      72, g, fi * 32);
            v16bf bk1 = load_frag_b(lk, 16 + ln, 72, g, fi * 32);
            s0 = __builtin_amdgcn_wmma_f32_16x16x32_bf16(false, qf[fi], false, bk0,
                                                         (short)0, s0, false, false);
            s1 = __builtin_amdgcn_wmma_f32_16x16x32_bf16(false, qf[fi], false, bk1,
                                                         (short)0, s1, false, false);
        }

        // online softmax across 16-lane half-wave row groups
        const int kc0 = kt + ln, kc1 = kt + 16 + ln;
        for (int r = 0; r < 8; ++r) {
            const int qi = qbase + r + (g << 3);
            float a0 = (kc0 <= qi) ? s0[r] : -1e30f;
            float a1 = (kc1 <= qi) ? s1[r] : -1e30f;
            float mloc = fmaxf(a0, a1);
            for (int off = 8; off; off >>= 1) mloc = fmaxf(mloc, __shfl_xor(mloc, off, 16));
            float mnew = fmaxf(Mr[r], mloc);
            float al = __expf(Mr[r] - mnew);
            float p0 = __expf(a0 - mnew);
            float p1 = __expf(a1 - mnew);
            float ls = p0 + p1;
            for (int off = 8; off; off >>= 1) ls += __shfl_xor(ls, off, 16);
            Lr[r] = Lr[r] * al + ls;
            Mr[r] = mnew;
            for (int t = 0; t < 4; ++t) oacc[t][r] *= al;
            int m = r + (g << 3);
            myP[m * 40 + ln]      = f2bf(p0);   // C-layout -> A-layout via LDS
            myP[m * 40 + 16 + ln] = f2bf(p1);
        }
        asm volatile("s_wait_dscnt 0" ::: "memory");  // same-wave LDS store->load handoff

        // O += P * V : A = P (16x32 keys), B = V^T [dh][key], 4 dh tiles
        v16bf pf = load_frag_a(myP, ln, 40, g, 0);
        for (int t = 0; t < 4; ++t) {
            v16bf vf = load_frag_b(lv, t * 16 + ln, 40, g, 0);
            oacc[t] = __builtin_amdgcn_wmma_f32_16x16x32_bf16(false, pf, false, vf,
                                                              (short)0, oacc[t], false, false);
        }
    }

    for (int r = 0; r < 8; ++r) {
        float inv = 1.0f / Lr[r];
        int qi = qbase + r + (g << 3);
        size_t rowoff = ((size_t)(b * 2048) + qi) * 1024 + (size_t)h * 64;
        for (int t = 0; t < 4; ++t)
            Ob[rowoff + t * 16 + ln] = f2bf(oacc[t][r] * inv);
    }
}

// ---------------------------------------------------------------- host launcher
extern "C" void kernel_launch(void* const* d_in, const int* in_sizes, int n_in,
                              void* d_out, int out_size, void* d_ws, size_t ws_size,
                              hipStream_t stream) {
    const float* x    = (const float*)d_in[0];   // [4,2048,1024]
    const float* wqkv = (const float*)d_in[1];   // [1024,3072]
    const float* wout = (const float*)d_in[2];   // [1024,1024]
    float* out = (float*)d_out;                  // [4,2048,1024] fp32

    unsigned short* xb     = (unsigned short*)d_ws;          // 8388608   x bf16 [8192,1024]
    unsigned short* wqkvT  = xb    + 8388608;                // 3145728   [3072,1024]
    unsigned short* woutT  = wqkvT + 3145728;                // 1048576   [1024,1024]
    unsigned short* qb     = woutT + 1048576;                // 8388608   [B,H,T,64]
    unsigned short* kb     = qb    + 8388608;                // 8388608   [B,H,T,64]
    unsigned short* vT     = kb    + 8388608;                // 8388608   [B,H,64,T]
    unsigned short* attnb  = vT    + 8388608;                // 8388608   [B*T, D]

    cvt_f32_bf16<<<8388608 / 2048, 256, 0, stream>>>(x, xb);
    cvt_transpose_bf16<<<dim3(3072 / 32, 1024 / 32), 256, 0, stream>>>(wqkv, wqkvT, 1024, 3072);
    cvt_transpose_bf16<<<dim3(1024 / 32, 1024 / 32), 256, 0, stream>>>(wout, woutT, 1024, 1024);

    gemm_bf16<<<dim3(3072 / 128, 8192 / 128), 256, 0, stream>>>(
        xb, wqkvT, nullptr, qb, kb, vT, 3072, 1024, 1);

    flash_attn<<<dim3(2048 / 128, 64), 256, 0, stream>>>(qb, kb, vT, attnb);

    gemm_bf16<<<dim3(1024 / 128, 8192 / 128), 256, 0, stream>>>(
        attnb, woutT, out, nullptr, nullptr, nullptr, 1024, 1024, 0);
}
